// ACAT_58866821759512
// MI455X (gfx1250) — matrix operand order repaired and checked by reading, compiled, not run on hardware
//
#include <hip/hip_runtime.h>

typedef __attribute__((ext_vector_type(2))) float v2f;
typedef __attribute__((ext_vector_type(8))) float v8f;
typedef __attribute__((ext_vector_type(8))) _Float16 v8h;
typedef __attribute__((ext_vector_type(16))) _Float16 v16h;

// D(16x16,f32) = A(16x4,f32) x B(4x16,f32) + C   -- V_WMMA_F32_16X16X4_F32
__device__ __forceinline__ v8f wmma_f32(v2f a, v2f b, v8f c) {
  return __builtin_amdgcn_wmma_f32_16x16x4_f32(false, a, false, b, (short)0, c,
                                               false, false);
}
// D(16x16,f32) = A(16x32,f16) x B(32x16,f16) + C -- V_WMMA_F32_16X16X32_F16
__device__ __forceinline__ v8f wmma_f16(v16h a, v16h b, v8f c) {
  return __builtin_amdgcn_wmma_f32_16x16x32_f16(false, a, false, b, (short)0, c,
                                                false, false);
}
__device__ __forceinline__ v16h cat8(v8h a, v8h b) {
  return __builtin_shufflevector(a, b, 0, 1, 2, 3, 4, 5, 6, 7, 8, 9, 10, 11, 12,
                                 13, 14, 15);
}

// ---------------------------------------------------------------------------
// Prep 1: split+transpose X: [4][512][1024] f32 -> hi/lo f16 [4][1024][512]
// ---------------------------------------------------------------------------
__global__ __launch_bounds__(256) void split_x_kernel(
    const float* __restrict__ X, _Float16* __restrict__ Hi,
    _Float16* __restrict__ Lo) {
  size_t i = (size_t)blockIdx.x * 256 + threadIdx.x;  // 4*1024*512 outputs
  int ci = (int)(i & 511);
  int x = (int)((i >> 9) & 1023);
  int bb = (int)(i >> 19);
  float v = X[((size_t)bb * 512 + ci) * 1024 + x];
  _Float16 h = (_Float16)v;
  Hi[i] = h;
  Lo[i] = (_Float16)(v - (float)h);
}

// ---------------------------------------------------------------------------
// Prep 2: split+transpose W: [512co][512ci][f] f32 -> hi/lo f16 [f][512co][512ci]
// ---------------------------------------------------------------------------
__global__ __launch_bounds__(256) void split_w_kernel(
    const float* __restrict__ W, _Float16* __restrict__ Hi,
    _Float16* __restrict__ Lo, int f, int total) {
  size_t i = (size_t)blockIdx.x * 256 + threadIdx.x;
  if (i >= (size_t)total) return;
  int ci = (int)(i & 511);
  int co = (int)((i >> 9) & 511);
  int j = (int)(i >> 18);
  float v = W[((size_t)co * 512 + ci) * f + j];
  _Float16 h = (_Float16)v;
  Hi[i] = h;
  Lo[i] = (_Float16)(v - (float)h);
}

// ---------------------------------------------------------------------------
// Kernel 1: Conv1d(512->512,'same') + bias, implicit GEMM on the f16 WMMA
// pipe with hi/lo split-precision (hi*hi + hi*lo + lo*hi ~ fp32 accuracy).
// Xt: [4][1024][512] (ci contiguous)   Wt: [f][512co][512ci]
// Block = 256 threads = 8 waves (4 co x 2 x); wave tile 16co x 16x.
// ---------------------------------------------------------------------------
__global__ __launch_bounds__(256) void conv_bias_kernel(
    const _Float16* __restrict__ XtHi, const _Float16* __restrict__ XtLo,
    const _Float16* __restrict__ WtHi, const _Float16* __restrict__ WtLo,
    const float* __restrict__ bias, float* __restrict__ Y, int f) {
  const int pad = f >> 1;
  const int lane = threadIdx.x & 31;
  const int wave = threadIdx.x >> 5;
  const int wm = wave >> 1, wn = wave & 1;
  const int bb = blockIdx.z;
  const int co0 = blockIdx.y * 64 + wm * 16;
  const int x0 = blockIdx.x * 32 + wn * 16;

  const int m = lane & 15;            // A row (co) / B col (x) / D col
  const int kh8 = (lane >> 4) * 8;    // unused; A fragment also uses kh16 (k = 16h + s)
  const int kh16 = (lane >> 4) * 16;  // B-fragment K base
  const int rbase = (lane >> 4) * 8;  // D rows r / r+8 per lane half

  v8f acc = {0.f, 0.f, 0.f, 0.f, 0.f, 0.f, 0.f, 0.f};
  const v16h zz = {0, 0, 0, 0, 0, 0, 0, 0, 0, 0, 0, 0, 0, 0, 0, 0};

  for (int ci0 = 0; ci0 < 512; ci0 += 32) {
    for (int j = 0; j < f; ++j) {
      const _Float16* wh = WtHi + ((size_t)j * 512 + co0 + m) * 512 + ci0;
      const _Float16* wl = WtLo + ((size_t)j * 512 + co0 + m) * 512 + ci0;
      v16h ahi = cat8(*(const v8h*)(wh + kh16), *(const v8h*)(wh + kh16 + 8));
      v16h alo = cat8(*(const v8h*)(wl + kh16), *(const v8h*)(wl + kh16 + 8));
      int x = x0 + m + j - pad;
      v16h bhi, blo;
      if ((unsigned)x < 1024u) {
        const _Float16* xh = XtHi + ((size_t)bb * 1024 + x) * 512 + ci0 + kh16;
        const _Float16* xl = XtLo + ((size_t)bb * 1024 + x) * 512 + ci0 + kh16;
        bhi = cat8(*(const v8h*)xh, *(const v8h*)(xh + 8));
        blo = cat8(*(const v8h*)xl, *(const v8h*)(xl + 8));
      } else {
        bhi = zz;
        blo = zz;
      }
      acc = wmma_f16(ahi, bhi, acc);
      acc = wmma_f16(ahi, blo, acc);
      acc = wmma_f16(alo, bhi, acc);
    }
  }
  float* Yb = Y + ((size_t)bb * 512 + co0) * 1024 + x0;
#pragma unroll
  for (int r = 0; r < 8; ++r) {
    int row = rbase + r;
    Yb[(size_t)row * 1024 + m] = acc[r] + bias[co0 + row];
  }
}

// ---------------------------------------------------------------------------
// Kernel 2: per-(slice,channel) batch-norm stats over (b=4, L=1024).
// Ycat: [6][4][512][1024]; stats: [6][512][2] = {mean, rsqrt(var+eps)}
// ---------------------------------------------------------------------------
__global__ __launch_bounds__(256) void bn_stats_kernel(
    const float* __restrict__ Y, float* __restrict__ stats) {
  const int c = blockIdx.x;
  const int s = blockIdx.y;
  const float* base = Y + (size_t)s * 2097152 + (size_t)c * 1024;
  float sum = 0.f, sq = 0.f;
  for (int bb = 0; bb < 4; ++bb) {
    const float* row = base + (size_t)bb * 524288;
    for (int x = threadIdx.x; x < 1024; x += 256) {
      float v = row[x];
      sum += v;
      sq += v * v;
    }
  }
  __shared__ float s1[256], s2[256];
  s1[threadIdx.x] = sum;
  s2[threadIdx.x] = sq;
  __syncthreads();
  for (int off = 128; off > 0; off >>= 1) {
    if (threadIdx.x < off) {
      s1[threadIdx.x] += s1[threadIdx.x + off];
      s2[threadIdx.x] += s2[threadIdx.x + off];
    }
    __syncthreads();
  }
  if (threadIdx.x == 0) {
    float mean = s1[0] * (1.f / 4096.f);
    float var = s2[0] * (1.f / 4096.f) - mean * mean;
    stats[(s * 512 + c) * 2 + 0] = mean;
    stats[(s * 512 + c) * 2 + 1] = rsqrtf(var + 1e-5f);
  }
}

// ---------------------------------------------------------------------------
// Kernel 3: BN apply + ELU over [6][4][512][1024], emitting hi/lo f16 split
// of the result (same flat layout == Q_p/K_p [4][8][3][1024][64]).
// ---------------------------------------------------------------------------
__global__ __launch_bounds__(256) void bn_elu_split_kernel(
    const float* __restrict__ Y, const float* __restrict__ stats,
    const float* __restrict__ gamma, const float* __restrict__ beta,
    _Float16* __restrict__ Hi, _Float16* __restrict__ Lo) {
  size_t i = (size_t)blockIdx.x * 256 + threadIdx.x;
  int c = (int)((i >> 10) & 511);
  int s = (int)(i >> 21);
  float mean = stats[(s * 512 + c) * 2 + 0];
  float rstd = stats[(s * 512 + c) * 2 + 1];
  float y = Y[i];
  y = (y - mean) * rstd * gamma[c] + beta[c];
  y = y > 0.f ? y : (expf(y) - 1.f);
  _Float16 h = (_Float16)y;
  Hi[i] = h;
  Lo[i] = (_Float16)(y - (float)h);
}

// ---------------------------------------------------------------------------
// Kernel 4: fused attention. Scores on the f16 WMMA pipe with hi/lo split
// (fp32-class accuracy); softmax + max-over-p exact f32; context on f32 WMMA.
// Q/K hi-lo: [4][8][3][1024][64] f16. V: [4][8][1024][64] f32.
// One workgroup per (b,h,16-row q-tile); 8 waves split k (128 cols each).
// ---------------------------------------------------------------------------
__global__ __launch_bounds__(256) void attn_kernel(
    const _Float16* __restrict__ QHi, const _Float16* __restrict__ QLo,
    const _Float16* __restrict__ KHi, const _Float16* __restrict__ KLo,
    const float* __restrict__ V, float* __restrict__ ctx_out,
    float* __restrict__ attn_out) {
  const int lane = threadIdx.x & 31;
  const int wave = threadIdx.x >> 5;
  const int m = lane & 15;
  const int khalf = (lane >> 4) * 2;  // f32-WMMA K base
  const int kh8 = (lane >> 4) * 8;    // unused; f16 A fragment also uses kh16
  const int kh16 = (lane >> 4) * 16;  // f16 B-fragment K base
  const int rbase = (lane >> 4) * 8;
  const int q0 = blockIdx.x * 16;
  const size_t bh = (size_t)blockIdx.z * 8 + blockIdx.y;

  __shared__ alignas(16) _Float16 Qsh[3][16][72];
  __shared__ alignas(16) _Float16 Qsl[3][16][72];
  __shared__ float At[8][16][17];
  __shared__ float WStats[8][3][16][2];
  __shared__ float Ms[3][16], Ssi[3][16];
  __shared__ float CtxS[8][16][68];

  for (int e = threadIdx.x; e < 3 * 16 * 64; e += 256) {
    int p = e >> 10, row = (e >> 6) & 15, d = e & 63;
    size_t gi = ((bh * 3 + p) * 1024 + (q0 + row)) * 64 + d;
    Qsh[p][row][d] = QHi[gi];
    Qsl[p][row][d] = QLo[gi];
  }
  __syncthreads();

  const int kstart = wave * 128;
  float mr[3][8], sr[3][8];
#pragma unroll
  for (int p = 0; p < 3; ++p)
#pragma unroll
    for (int r = 0; r < 8; ++r) {
      mr[p][r] = -1e30f;
      sr[p][r] = 0.f;
    }

  // ---- pass 1: online (max, sum(exp)) over this wave's k range ----
  for (int kt = 0; kt < 8; ++kt) {
    const int k0 = kstart + kt * 16;
#pragma unroll
    for (int p = 0; p < 3; ++p) {
      const _Float16* Kh = KHi + (((bh * 3 + p) * 1024 + k0 + m) * 64);
      const _Float16* Kl = KLo + (((bh * 3 + p) * 1024 + k0 + m) * 64);
      v8f sc = {0.f, 0.f, 0.f, 0.f, 0.f, 0.f, 0.f, 0.f};
#pragma unroll
      for (int dc = 0; dc < 2; ++dc) {
        const int db = dc * 32;
        v16h ah = cat8(*(const v8h*)&Qsh[p][m][db + kh16],
                       *(const v8h*)&Qsh[p][m][db + kh16 + 8]);
        v16h al = cat8(*(const v8h*)&Qsl[p][m][db + kh16],
                       *(const v8h*)&Qsl[p][m][db + kh16 + 8]);
        v16h bhh = cat8(*(const v8h*)(Kh + db + kh16),
                        *(const v8h*)(Kh + db + kh16 + 8));
        v16h bll = cat8(*(const v8h*)(Kl + db + kh16),
                        *(const v8h*)(Kl + db + kh16 + 8));
        sc = wmma_f16(ah, bhh, sc);
        sc = wmma_f16(ah, bll, sc);
        sc = wmma_f16(al, bhh, sc);
      }
#pragma unroll
      for (int r = 0; r < 8; ++r) {
        float x = sc[r] * 0.125f;  // 1/sqrt(64)
        float mx = x;
        mx = fmaxf(mx, __shfl_xor(mx, 1, 32));
        mx = fmaxf(mx, __shfl_xor(mx, 2, 32));
        mx = fmaxf(mx, __shfl_xor(mx, 4, 32));
        mx = fmaxf(mx, __shfl_xor(mx, 8, 32));
        float mnew = fmaxf(mr[p][r], mx);
        float e = __expf(x - mnew);
        e += __shfl_xor(e, 1, 32);
        e += __shfl_xor(e, 2, 32);
        e += __shfl_xor(e, 4, 32);
        e += __shfl_xor(e, 8, 32);
        sr[p][r] = sr[p][r] * __expf(mr[p][r] - mnew) + e;
        mr[p][r] = mnew;
      }
    }
  }
  if (m == 0) {
#pragma unroll
    for (int p = 0; p < 3; ++p)
#pragma unroll
      for (int r = 0; r < 8; ++r) {
        WStats[wave][p][rbase + r][0] = mr[p][r];
        WStats[wave][p][rbase + r][1] = sr[p][r];
      }
  }
  __syncthreads();
  if (threadIdx.x < 48) {
    int p = threadIdx.x >> 4, row = threadIdx.x & 15;
    float M = -1e30f;
    for (int w = 0; w < 8; ++w) M = fmaxf(M, WStats[w][p][row][0]);
    float S = 0.f;
    for (int w = 0; w < 8; ++w)
      S += WStats[w][p][row][1] * __expf(WStats[w][p][row][0] - M);
    Ms[p][row] = M;
    Ssi[p][row] = 1.f / S;
  }
  __syncthreads();
#pragma unroll
  for (int p = 0; p < 3; ++p)
#pragma unroll
    for (int r = 0; r < 8; ++r) {
      mr[p][r] = Ms[p][rbase + r];
      sr[p][r] = Ssi[p][rbase + r];
    }

  // ---- pass 2: recompute scores, emit attn, accumulate context ----
  const float* Vb = V + bh * (1024 * 64);
  float* Ao = attn_out + (bh * 1024 + q0) * 1024;
  v8f ctx[4];
#pragma unroll
  for (int nt = 0; nt < 4; ++nt) {
    v8f z = {0.f, 0.f, 0.f, 0.f, 0.f, 0.f, 0.f, 0.f};
    ctx[nt] = z;
  }

  for (int kt = 0; kt < 8; ++kt) {
    const int k0 = kstart + kt * 16;
    v8f sc[3];
#pragma unroll
    for (int p = 0; p < 3; ++p) {
      const _Float16* Kh = KHi + (((bh * 3 + p) * 1024 + k0 + m) * 64);
      const _Float16* Kl = KLo + (((bh * 3 + p) * 1024 + k0 + m) * 64);
      v8f z = {0.f, 0.f, 0.f, 0.f, 0.f, 0.f, 0.f, 0.f};
      sc[p] = z;
#pragma unroll
      for (int dc = 0; dc < 2; ++dc) {
        const int db = dc * 32;
        v16h ah = cat8(*(const v8h*)&Qsh[p][m][db + kh16],
                       *(const v8h*)&Qsh[p][m][db + kh16 + 8]);
        v16h al = cat8(*(const v8h*)&Qsl[p][m][db + kh16],
                       *(const v8h*)&Qsl[p][m][db + kh16 + 8]);
        v16h bhh = cat8(*(const v8h*)(Kh + db + kh16),
                        *(const v8h*)(Kh + db + kh16 + 8));
        v16h bll = cat8(*(const v8h*)(Kl + db + kh16),
                        *(const v8h*)(Kl + db + kh16 + 8));
        sc[p] = wmma_f16(ah, bhh, sc[p]);
        sc[p] = wmma_f16(ah, bll, sc[p]);
        sc[p] = wmma_f16(al, bhh, sc[p]);
      }
    }
    __syncthreads();  // previous iteration's At reads are done
#pragma unroll
    for (int r = 0; r < 8; ++r) {
      int row = rbase + r;
      float a0 = __expf(sc[0][r] * 0.125f - mr[0][r]) * sr[0][r];
      float a1 = __expf(sc[1][r] * 0.125f - mr[1][r]) * sr[1][r];
      float a2 = __expf(sc[2][r] * 0.125f - mr[2][r]) * sr[2][r];
      float av = fmaxf(a0, fmaxf(a1, a2));
      At[wave][row][m] = av;
      Ao[(size_t)row * 1024 + k0 + m] = av;
    }
    __syncthreads();  // At visible to all lanes of this wave
#pragma unroll
    for (int nt = 0; nt < 4; ++nt) {
      const int d0 = nt * 16;
#pragma unroll
      for (int ks = 0; ks < 4; ++ks) {
        v2f a, b;
        a.x = At[wave][m][ks * 4 + khalf];
        a.y = At[wave][m][ks * 4 + khalf + 1];
        b.x = Vb[(size_t)(k0 + ks * 4 + khalf + 0) * 64 + d0 + m];
        b.y = Vb[(size_t)(k0 + ks * 4 + khalf + 1) * 64 + d0 + m];
        ctx[nt] = wmma_f32(a, b, ctx[nt]);
      }
    }
  }
#pragma unroll
  for (int nt = 0; nt < 4; ++nt)
#pragma unroll
    for (int r = 0; r < 8; ++r) CtxS[wave][rbase + r][nt * 16 + m] = ctx[nt][r];
  __syncthreads();
  for (int e = threadIdx.x; e < 16 * 64; e += 256) {
    int row = e >> 6, d = e & 63;
    float s = 0.f;
    for (int w = 0; w < 8; ++w) s += CtxS[w][row][d];
    ctx_out[(bh * 1024 + q0 + row) * 64 + d] = s;
  }
}

// ---------------------------------------------------------------------------
extern "C" void kernel_launch(void* const* d_in, const int* in_sizes, int n_in,
                              void* d_out, int out_size, void* d_ws,
                              size_t ws_size, hipStream_t stream) {
  (void)in_sizes; (void)n_in; (void)out_size; (void)ws_size;
  const float* Q = (const float*)d_in[0];
  const float* K = (const float*)d_in[1];
  const float* V = (const float*)d_in[2];
  const float* wq[3] = {(const float*)d_in[3], (const float*)d_in[7],
                        (const float*)d_in[11]};
  const float* bq[3] = {(const float*)d_in[4], (const float*)d_in[8],
                        (const float*)d_in[12]};
  const float* wk[3] = {(const float*)d_in[5], (const float*)d_in[9],
                        (const float*)d_in[13]};
  const float* bk[3] = {(const float*)d_in[6], (const float*)d_in[10],
                        (const float*)d_in[14]};
  const float* gamma = (const float*)d_in[15];
  const float* beta = (const float*)d_in[16];

  // Workspace:
  //   Yq[3][4][512][1024] f32 | Yk f32 | stats[6][512][2] f32 |
  //   prep region (dead after convs): XtQ/XtK hi+lo f16, Wt hi+lo f16
  //   attn region (overlays prep): Hi[2][3*2097152] f16, Lo[...] f16
  float* Yq = (float*)d_ws;
  float* Yk = Yq + 3ull * 2097152;
  float* stats = Yk + 3ull * 2097152;
  _Float16* XtQHi = (_Float16*)(stats + 6144);
  _Float16* XtQLo = XtQHi + 2097152;
  _Float16* XtKHi = XtQLo + 2097152;
  _Float16* XtKLo = XtKHi + 2097152;
  _Float16* WtHi = XtKLo + 2097152;  // sized for f=9 (reused per conv)
  _Float16* WtLo = WtHi + 2359296;
  // attn-time f16 buffers: overlay the prep region (safe: stream-ordered)
  _Float16* AHi = XtQHi;             // 12582912 elems: [Qp | Kp]
  _Float16* ALo = AHi + 12582912;

  split_x_kernel<<<dim3(8192), 256, 0, stream>>>(Q, XtQHi, XtQLo);
  split_x_kernel<<<dim3(8192), 256, 0, stream>>>(K, XtKHi, XtKLo);

  const int fs[3] = {3, 7, 9};
  dim3 cgrid(32, 8, 4), cblk(256);
  for (int i = 0; i < 3; ++i) {
    int f = fs[i];
    int wtot = 512 * 512 * f;
    dim3 wgrid((wtot + 255) / 256);
    split_w_kernel<<<wgrid, 256, 0, stream>>>(wq[i], WtHi, WtLo, f, wtot);
    conv_bias_kernel<<<cgrid, cblk, 0, stream>>>(
        XtQHi, XtQLo, WtHi, WtLo, bq[i], Yq + (size_t)i * 2097152, f);
    split_w_kernel<<<wgrid, 256, 0, stream>>>(wk[i], WtHi, WtLo, f, wtot);
    conv_bias_kernel<<<cgrid, cblk, 0, stream>>>(
        XtKHi, XtKLo, WtHi, WtLo, bk[i], Yk + (size_t)i * 2097152, f);
  }
  bn_stats_kernel<<<dim3(512, 6), 256, 0, stream>>>(Yq, stats);
  // consumes prep region's liveness: from here on AHi/ALo own that memory
  bn_elu_split_kernel<<<dim3(49152), 256, 0, stream>>>(Yq, stats, gamma, beta,
                                                       AHi, ALo);

  float* ctx_out = (float*)d_out;       // [4][8][1024][64]
  float* attn_out = ctx_out + 2097152;  // [4][8][1024][1024]
  attn_kernel<<<dim3(64, 8, 4), 256, 0, stream>>>(
      AHi, ALo, AHi + 6291456, ALo + 6291456, V, ctx_out, attn_out);
}
